// FMHA_35862976922019
// MI455X (gfx1250) — matrix operand order repaired and checked by reading, not compiled, not run on hardware
//
#include <hip/hip_runtime.h>

// ---------------------------------------------------------------------------
// Problem constants (match the JAX reference)
// ---------------------------------------------------------------------------
#define D_MODEL 512
#define H_HEADS 8
#define LXG     32
#define LYG     32
#define KXK     5
#define KYK     5
#define NP      (LXG * LYG)   // 1024
#define BATCH   64
#define M_ROWS  (BATCH * NP)  // 65536

typedef __attribute__((ext_vector_type(16))) __bf16 v16bf;
typedef __attribute__((ext_vector_type(8)))  __bf16 v8bf;
typedef __attribute__((ext_vector_type(8)))  float  v8f;

// ---------------------------------------------------------------------------
// GEMM + bias:  C[M,N] = A[M,K] * W[K,N] + bias[N]    (K = N = 512)
//
// Block: 256 threads = 8 waves. Block tile = 128(M) x 64(N).
// Each wave owns a 16(M) x 64(N) strip = 4 independent 16x16 WMMA
// accumulators, so one A fragment feeds 4 v_wmma_f32_16x16x32_bf16 (4x A
// reuse, and the 4 back-to-back WMMAs have no D->C RAW hazard).
// The 512K x 64N f32 weight panel is converted to bf16 and staged in LDS
// once (64 KB), then read back as ds_load_b128 fragments.
// ---------------------------------------------------------------------------
template <bool A_IS_F32, bool OUT_F32>
__global__ __launch_bounds__(256) void gemm_bias_kernel(
    const void* __restrict__ A_, const float* __restrict__ W,
    const float* __restrict__ bias, void* __restrict__ C_)
{
  // LDS B tile: [n][k] layout, n-stride = 512 bf16 (1 KB row) -> each B
  // fragment is 2 runs of 8 contiguous K values = two 16-byte LDS reads.
  __shared__ __align__(32) __bf16 Bs[64 * D_MODEL];   // 64 KB

  const int tid  = threadIdx.x;
  const int lane = tid & 31;
  const int wave = tid >> 5;
  const int n0   = blockIdx.y * 64;
  const int m0   = blockIdx.x * 128 + wave * 16;

  // Cooperative stage of B panel (f32 -> bf16). 32768 elems / 256 thr.
  for (int i = tid; i < 64 * D_MODEL; i += 256) {
    const int n = i >> 9;              // i / 512
    const int k = i & (D_MODEL - 1);   // i % 512
    Bs[i] = (__bf16)W[(size_t)k * D_MODEL + (n0 + n)];
  }
  __syncthreads();

  // ISA 16-bit A layout (16x32 MxK): lanes 0-15 / 16-31 carry M=0..15 with
  // K groups {0..7,16..23} / {8..15,24..31}; elements 2v,2v+1 live in VGPR v.
  const int m     = m0 + (lane & 15);
  const int khalf = lane >> 4;
  const int nb    = lane & 15;

  v8f acc[4] = {v8f{}, v8f{}, v8f{}, v8f{}};

#pragma unroll
  for (int k0 = 0; k0 < D_MODEL; k0 += 32) {
    const int kA0 = k0 + khalf * 8;        // first 8 contiguous K
    const int kA1 = k0 + 16 + khalf * 8;   // second 8 contiguous K

    v16bf a;
    if constexpr (A_IS_F32) {
      const float* Arow = (const float*)A_ + (size_t)m * D_MODEL;
      if (k0 + 32 < D_MODEL)
        __builtin_prefetch(Arow + kA0 + 32, 0, 3);   // global_prefetch, near
      float tmp[16];
      *(float4*)&tmp[0]  = *(const float4*)(Arow + kA0);
      *(float4*)&tmp[4]  = *(const float4*)(Arow + kA0 + 4);
      *(float4*)&tmp[8]  = *(const float4*)(Arow + kA1);
      *(float4*)&tmp[12] = *(const float4*)(Arow + kA1 + 4);
#pragma unroll
      for (int e = 0; e < 16; ++e) a[e] = (__bf16)tmp[e];
    } else {
      const __bf16* Arow = (const __bf16*)A_ + (size_t)m * D_MODEL;
      if (k0 + 32 < D_MODEL)
        __builtin_prefetch(Arow + kA0 + 32, 0, 3);
      const v8bf lo = *(const v8bf*)(Arow + kA0);
      const v8bf hi = *(const v8bf*)(Arow + kA1);
#pragma unroll
      for (int e = 0; e < 8; ++e) { a[e] = lo[e]; a[8 + e] = hi[e]; }
    }

    // 4 N-tiles share this A fragment; independent accumulators pipeline
    // through the XDL with no WMMA->WMMA RAW hazard.
#pragma unroll
    for (int j = 0; j < 4; ++j) {
      // ISA 16-bit B layout (32x16 KxN): lane%16 = N, half-wave selects the
      // K groups {0..7,16..23} / {8..15,24..31}, matching A.
      const v8bf blo =
          *(const v8bf*)(&Bs[(j * 16 + nb) * D_MODEL + kA0]);
      const v8bf bhi =
          *(const v8bf*)(&Bs[(j * 16 + nb) * D_MODEL + kA1]);
      v16bf b;
#pragma unroll
      for (int e = 0; e < 8; ++e) { b[e] = blo[e]; b[8 + e] = bhi[e]; }
      acc[j] = __builtin_amdgcn_wmma_f32_16x16x32_bf16(
          /*neg_a=*/false, a, /*neg_b=*/false, b,
          /*c_mod=*/(short)0, acc[j], /*reuse_a=*/false, /*reuse_b=*/false);
    }
  }

  // C/D layout: lane%16 = N; half-wave selects M block of 8; VGPR r = M row.
  const int mBase = m0 + (lane >> 4) * 8;
#pragma unroll
  for (int j = 0; j < 4; ++j) {
    const int   nOut = n0 + j * 16 + nb;
    const float bv   = bias[nOut];
#pragma unroll
    for (int r = 0; r < 8; ++r) {
      const float  val = acc[j][r] + bv;
      const size_t off = (size_t)(mBase + r) * D_MODEL + nOut;
      if constexpr (OUT_F32) ((float*)C_)[off]  = val;
      else                   ((__bf16*)C_)[off] = (__bf16)val;
    }
  }
}

// ---------------------------------------------------------------------------
// Per-head periodic 5x5 mixing (the "J" einsum is a circular depthwise conv
// on the 32x32 torus):
//   attn[b,(sx,sy),h,d] = sum_{kx,ky} Jk[h,kx*5+ky] *
//                         v[b, ((sx-2+kx)%32, (sy-2+ky)%32), h, d]
// v (64 MB bf16) is L2-resident (192 MB L2), so the 25 gathers are L2 hits.
// ---------------------------------------------------------------------------
__global__ __launch_bounds__(512) void conv_mix_kernel(
    const __bf16* __restrict__ v, const float* __restrict__ Jk,
    __bf16* __restrict__ attn)
{
  const int d  = threadIdx.x;      // 0..511
  const int p  = blockIdx.x;       // 0..1023
  const int bb = blockIdx.y;       // 0..63
  const int h  = d >> 6;           // head = d / 64
  const int sx = p >> 5;
  const int sy = p & 31;

  float jk[KXK * KYK];
#pragma unroll
  for (int t = 0; t < KXK * KYK; ++t) jk[t] = Jk[h * (KXK * KYK) + t];

  const __bf16* vb = v + (size_t)bb * NP * D_MODEL;

  float acc = 0.0f;
#pragma unroll
  for (int kx = 0; kx < KXK; ++kx) {
    const int qx = (sx + (LXG - KXK / 2) + kx) & (LXG - 1);
#pragma unroll
    for (int ky = 0; ky < KYK; ++ky) {
      const int qy = (sy + (LYG - KYK / 2) + ky) & (LYG - 1);
      const int q  = (qx << 5) | qy;
      acc += jk[kx * KYK + ky] * (float)vb[(size_t)q * D_MODEL + d];
    }
  }
  attn[((size_t)bb * NP + p) * D_MODEL + d] = (__bf16)acc;
}

// ---------------------------------------------------------------------------
// Launch:  x --(GEMM1: bf16 WMMA)--> v_ws --(conv)--> attn_ws
//            --(GEMM2: bf16 WMMA)--> out (f32)
// Workspace: v_ws (64 MB bf16) + attn_ws (64 MB bf16); both fully written
// before being read, every launch -> deterministic, poison-safe.
// ---------------------------------------------------------------------------
extern "C" void kernel_launch(void* const* d_in, const int* in_sizes, int n_in,
                              void* d_out, int out_size, void* d_ws, size_t ws_size,
                              hipStream_t stream) {
  (void)in_sizes; (void)n_in; (void)out_size; (void)ws_size;

  const float* x  = (const float*)d_in[0];  // (B, Np, 512) f32
  const float* vk = (const float*)d_in[1];  // (512, 512)   f32
  const float* vb = (const float*)d_in[2];  // (512,)       f32
  const float* Jk = (const float*)d_in[3];  // (8, 25)      f32
  const float* Wk = (const float*)d_in[4];  // (512, 512)   f32
  const float* Wb = (const float*)d_in[5];  // (512,)       f32
  float* out = (float*)d_out;               // (B, Np, 512) f32

  __bf16* v_ws    = (__bf16*)d_ws;
  __bf16* attn_ws = v_ws + (size_t)M_ROWS * D_MODEL;   // +64 MB

  const dim3 gGemm(M_ROWS / 128, D_MODEL / 64);        // (512, 8)
  const dim3 gConv(NP, BATCH);                         // (1024, 64)

  // v = x @ V + bv                (f32 A -> bf16 WMMA -> bf16 out)
  gemm_bias_kernel<true, false><<<gGemm, 256, 0, stream>>>(x, vk, vb, v_ws);
  // attn = circular 5x5 head-mix  (bf16 -> bf16)
  conv_mix_kernel<<<gConv, 512, 0, stream>>>(v_ws, Jk, attn_ws);
  // out = attn @ W + bw           (bf16 A -> bf16 WMMA -> f32 out)
  gemm_bias_kernel<false, true><<<gGemm, 256, 0, stream>>>(attn_ws, Wk, Wb, out);
}
